// Decoder_8950711845817
// MI455X (gfx1250) — matrix operand order repaired and checked by reading, not compiled, not run on hardware
//
#include <hip/hip_runtime.h>
#include <math.h>

#define DEV __device__ __forceinline__

typedef _Float16 h16;
typedef __attribute__((ext_vector_type(16))) _Float16 v16h;
typedef __attribute__((ext_vector_type(8)))  float    v8f;
typedef __attribute__((ext_vector_type(4)))  unsigned int u32x4;

union F16Frag { v16h v; h16 e[16]; u32x4 q[2]; };

// ---------------------------------------------------------------- constants
#define LSEQ   300
#define LPAD   304      // 19 tiles of 16
#define EMB    64
#define HEADS  8
#define HDIM   8
#define MLPH   256
#define QKVW   192
#define KVROWS 320      // qkv rows padded to 320 so attn*V K-loop (10x32) stays in-bounds
#define PSTRIDE 320     // prob strip K padded to 10*32

// ---------------------------------------------------------------- WMMA core
DEV v8f wmma16(v16h a, v16h b, v8f c) {
  // D = A(16x32 f16) * B(32x16 f16) + C(16x16 f32)
  return __builtin_amdgcn_wmma_f32_16x16x32_f16(false, a, false, b, (short)0, c, false, false);
}

// A fragment: rowbase points at A[row_m][k0]; K pairs contiguous in memory.
DEV v16h load_afrag(const h16* rowbase, int hi) {
  F16Frag f;
  f.q[0] = *(const u32x4*)(rowbase + hi * 16);       // dwords 0-3 : K = hi*16 + 0..7
  f.q[1] = *(const u32x4*)(rowbase + hi * 16 + 8);   // dwords 4-7 : K = hi*16 + 8..15
  return f.v;
}

// B fragment from transposed weights Wt[n][k]: rowbase = Wt + n*K + k0.
DEV v16h load_bfrag(const h16* rowbase, int hi) {
  F16Frag f;
  f.q[0] = *(const u32x4*)(rowbase + hi * 16);       // dwords 0-3 : K = hi*16 + 0..7
  f.q[1] = *(const u32x4*)(rowbase + hi * 16 + 8);   // dwords 4-7 : K = hi*16 + 8..15
  return f.v;
}

DEV float gelu_f(float x) {
  float x3 = x * x * x;
  return 0.5f * x * (1.f + tanhf(0.7978845608f * (x + 0.044715f * x3)));
}
DEV float sigmoid_f(float x) { return 1.f / (1.f + __expf(-x)); }

// ---------------------------------------------------------------- generic GEMM
// C[304 x (ntiles*16)] = act( A[304 x K](f16) * Wt^T + bias ), tiles spread over 8 waves.
// ACT: 0 none, 1 gelu, 2 sigmoid.  OUT: 0 store f16 to C16, 1 accumulate f32 into C32.
template <int ACT, int OUT>
DEV void wave_gemm(const h16* A, int lda, const h16* Wt, int K,
                   const float* bias, int ntiles,
                   h16* C16, int ldc, float* C32, int ldc32) {
  const int tid = threadIdx.x;
  const int w = tid >> 5, l = tid & 31, hi = l >> 4, lan = l & 15;
  const int total = 19 * ntiles;
  for (int t = w; t < total; t += 8) {
    const int rt = t / ntiles, ct = t - rt * ntiles;
    v8f c = {0.f, 0.f, 0.f, 0.f, 0.f, 0.f, 0.f, 0.f};
    for (int ks = 0; ks < K; ks += 32) {
      v16h a = load_afrag(A + (rt * 16 + lan) * lda + ks, hi);
      v16h b = load_bfrag(Wt + (ct * 16 + lan) * K + ks, hi);
      c = wmma16(a, b, c);
    }
    const float bb = bias[ct * 16 + lan];
#pragma unroll
    for (int v = 0; v < 8; ++v) {
      float val = c[v] + bb;
      if (ACT == 1) val = gelu_f(val);
      if (ACT == 2) val = sigmoid_f(val);
      const int row = rt * 16 + v + hi * 8;
      const int col = ct * 16 + lan;
      if (OUT == 0) C16[row * ldc + col] = (h16)val;
      else          C32[row * ldc32 + col] += val;
    }
  }
}

// Stage W[K][N] (f32, global) -> Wt[N][K] (f16, LDS)
DEV void load_wt(const float* __restrict__ W, int K, int N, h16* Wt) {
  for (int i = threadIdx.x; i < K * N; i += 256) {
    int k = i / N, n = i - k * N;
    Wt[n * K + k] = (h16)W[i];
  }
}

// LayerNorm rows of sH (f32) -> f16 dst; pad rows zeroed.
DEV void layernorm_f16(const float* S, const float* __restrict__ g,
                       const float* __restrict__ b, h16* D) {
  for (int t = threadIdx.x; t < LPAD; t += 256) {
    if (t < LSEQ) {
      const float* r = S + t * EMB;
      float mu = 0.f;
#pragma unroll
      for (int e = 0; e < EMB; ++e) mu += r[e];
      mu *= (1.f / 64.f);
      float var = 0.f;
#pragma unroll
      for (int e = 0; e < EMB; ++e) { float d = r[e] - mu; var += d * d; }
      var *= (1.f / 64.f);
      float rs = rsqrtf(var + 1e-5f);
#pragma unroll
      for (int e = 0; e < EMB; ++e)
        D[t * EMB + e] = (h16)((r[e] - mu) * rs * g[e] + b[e]);
    } else {
#pragma unroll
      for (int e = 0; e < EMB; ++e) D[t * EMB + e] = (h16)0.f;
    }
  }
}

// ---------------------------------------------------------------- attention
// qkv: [KVROWS][192] f16 (q|k|v, per-head blocks of 8). probs: 8 strips of [16][320] f16.
// o:   [LPAD][64] f16 output.  One wave == one head.
DEV void attention(const h16* qkv, h16* probs, h16* o) {
  const int tid = threadIdx.x;
  const int w = tid >> 5, l = tid & 31, hi = l >> 4, lan = l & 15;
  const int h = w;                      // 8 waves <-> 8 heads
  h16* myP = probs + w * (16 * PSTRIDE);
  const u32x4 z4 = {0u, 0u, 0u, 0u};

  // zero pad columns 304..319 of this wave's strip
  for (int i = l; i < 16 * 16; i += 32) {
    int r = i >> 4, c = i & 15;
    myP[r * PSTRIDE + 304 + c] = (h16)0.f;
  }

  for (int qt = 0; qt < 19; ++qt) {
    // ---- scores: Q(16x8, K-padded to 32) x K^T --------------------------
    F16Frag qa;
    {
      u32x4 t = *(const u32x4*)(qkv + (qt * 16 + lan) * QKVW + h * HDIM);
      qa.q[0] = hi ? z4 : t;            // K>=8 contributes zero
      qa.q[1] = z4;
    }
    v8f s[19];
#pragma unroll
    for (int kt = 0; kt < 19; ++kt) {
      F16Frag kb;
      u32x4 t = *(const u32x4*)(qkv + (kt * 16 + lan) * QKVW + 64 + h * HDIM);
      kb.q[0] = hi ? z4 : t;
      kb.q[1] = z4;
      v8f c = {0.f, 0.f, 0.f, 0.f, 0.f, 0.f, 0.f, 0.f};
      s[kt] = wmma16(qa.v, kb.v, c);
    }
    // ---- softmax over 304 keys (mask 300..303) --------------------------
    const float sc = 0.35355339059327f; // 1/sqrt(8)
    float rm[8];
#pragma unroll
    for (int v = 0; v < 8; ++v) rm[v] = -1e30f;
#pragma unroll
    for (int kt = 0; kt < 19; ++kt) {
#pragma unroll
      for (int v = 0; v < 8; ++v) {
        float val = s[kt][v] * sc;
        if (kt == 18 && lan >= 12) val = -1e30f;  // keys 300..303 masked
        s[kt][v] = val;
        rm[v] = fmaxf(rm[v], val);
      }
    }
#pragma unroll
    for (int v = 0; v < 8; ++v) {     // row max across the 16-lane half
      float m = rm[v];
      m = fmaxf(m, __shfl_xor(m, 1, 32));
      m = fmaxf(m, __shfl_xor(m, 2, 32));
      m = fmaxf(m, __shfl_xor(m, 4, 32));
      m = fmaxf(m, __shfl_xor(m, 8, 32));
      rm[v] = m;
    }
    float rs[8];
#pragma unroll
    for (int v = 0; v < 8; ++v) rs[v] = 0.f;
#pragma unroll
    for (int kt = 0; kt < 19; ++kt) {
#pragma unroll
      for (int v = 0; v < 8; ++v) {
        float e = __expf(s[kt][v] - rm[v]);
        s[kt][v] = e;
        rs[v] += e;
      }
    }
#pragma unroll
    for (int v = 0; v < 8; ++v) {
      float m = rs[v];
      m += __shfl_xor(m, 1, 32);
      m += __shfl_xor(m, 2, 32);
      m += __shfl_xor(m, 4, 32);
      m += __shfl_xor(m, 8, 32);
      rs[v] = 1.f / m;
    }
    // spill probabilities (f16) to the wave-private strip
#pragma unroll
    for (int kt = 0; kt < 19; ++kt) {
#pragma unroll
      for (int v = 0; v < 8; ++v)
        myP[(v + hi * 8) * PSTRIDE + kt * 16 + lan] = (h16)(s[kt][v] * rs[v]);
    }
    // ---- O = P(16x320) x V(320x8 padded to 16) --------------------------
    v8f oc = {0.f, 0.f, 0.f, 0.f, 0.f, 0.f, 0.f, 0.f};
    for (int ks = 0; ks < 10; ++ks) {
      v16h a = load_afrag(myP + lan * PSTRIDE + ks * 32, hi);
      F16Frag vb;
      const int col = 128 + h * HDIM + (lan & 7);  // lanes 8-15 duplicate col 0-7 (unused)
#pragma unroll
      for (int v = 0; v < 8; ++v) {
        int k = ks * 32 + 2 * v + hi * 16;
        vb.e[2 * v]     = qkv[k * QKVW + col];
        vb.e[2 * v + 1] = qkv[(k + 1) * QKVW + col];
      }
      oc = wmma16(a, vb.v, oc);
    }
    if (lan < 8) {
#pragma unroll
      for (int v = 0; v < 8; ++v)
        o[(qt * 16 + v + hi * 8) * EMB + h * HDIM + lan] = (h16)oc[v];
    }
  }
}

// ---------------------------------------------------------------- parameters
struct BlockW {
  const float *ln1_g, *ln1_b, *qkv_w, *qkv_b, *proj_w, *proj_b;
  const float *ln2_g, *ln2_b, *mlp1_w, *mlp1_b, *mlp2_w, *mlp2_b;
};
struct NetW {
  const float *x, *fc_w, *fc_b, *fcln_g, *fcln_b, *pos;
  const float *dec1_w, *dec1_b, *dec2_w, *dec2_b;
  const float *conv1_w, *conv1_b, *conv2_w, *conv2_b;
  BlockW blk[3];
};

// ---------------------------------------------------------------- mega kernel
__global__ __launch_bounds__(256) void decoder_fused(NetW P, float* __restrict__ out) {
  // LDS: 77,824 + 38,912 + 204,800 = 321,536 B (<= 320 KB/WGP)
  __shared__ float sH[LPAD * EMB];          // residual h (f32)
  __shared__ h16   sX[LPAD * EMB];          // f16 activations (LN out / attn out)
  __shared__ h16   sBig[102400];            // qkv | probs | Wt | mlp-hidden (time-shared)

  h16* sQKV  = sBig;                        // [320][192]           (61,440 h)
  h16* sScr  = sBig + KVROWS * QKVW;        // probs 8x[16][320] or Wt (40,960 h)
  h16* sM    = sBig;                        // mlp hidden [304][256](77,824 h)
  h16* sWt2  = sBig + LPAD * MLPH;          // Wt during mlp phases (24,576 h free)

  const int b   = blockIdx.x;
  const int tid = threadIdx.x;

  // ---------------- embed: x @ fc_w + fc_b -> LN -> + pos_embed ----------
  for (int t = tid; t < LPAD; t += 256) {
    if (t < LSEQ) {
      const float* xr = P.x + ((long)b * LSEQ + t) * 24;
      float xv[24];
#pragma unroll
      for (int k = 0; k < 24; ++k) xv[k] = xr[k];
      float y[EMB];
#pragma unroll
      for (int e = 0; e < EMB; ++e) {
        float acc = P.fc_b[e];
#pragma unroll
        for (int k = 0; k < 24; ++k) acc += xv[k] * P.fc_w[k * EMB + e];
        y[e] = acc;
      }
      float mu = 0.f;
#pragma unroll
      for (int e = 0; e < EMB; ++e) mu += y[e];
      mu *= (1.f / 64.f);
      float var = 0.f;
#pragma unroll
      for (int e = 0; e < EMB; ++e) { float d = y[e] - mu; var += d * d; }
      var *= (1.f / 64.f);
      float rs = rsqrtf(var + 1e-5f);
#pragma unroll
      for (int e = 0; e < EMB; ++e)
        sH[t * EMB + e] = (y[e] - mu) * rs * P.fcln_g[e] + P.fcln_b[e] + P.pos[t * EMB + e];
    } else {
#pragma unroll
      for (int e = 0; e < EMB; ++e) sH[t * EMB + e] = 0.f;
    }
  }
  __syncthreads();

  // ---------------- 3 transformer blocks ---------------------------------
#pragma unroll 1
  for (int blk = 0; blk < 3; ++blk) {
    const BlockW bw = P.blk[blk];

    layernorm_f16(sH, bw.ln1_g, bw.ln1_b, sX);
    __syncthreads();
    load_wt(bw.qkv_w, 64, 192, sScr);
    __syncthreads();
    wave_gemm<0, 0>(sX, EMB, sScr, 64, bw.qkv_b, 12, sQKV, QKVW, nullptr, 0);
    for (int i = tid; i < 16 * QKVW; i += 256) sQKV[304 * QKVW + i] = (h16)0.f;
    __syncthreads();

    attention(sQKV, sScr, sX);
    __syncthreads();

    load_wt(bw.proj_w, 64, 64, sScr);
    __syncthreads();
    wave_gemm<0, 1>(sX, EMB, sScr, 64, bw.proj_b, 4, nullptr, 0, sH, EMB);
    __syncthreads();

    layernorm_f16(sH, bw.ln2_g, bw.ln2_b, sX);
    __syncthreads();
    load_wt(bw.mlp1_w, 64, 256, sWt2);
    __syncthreads();
    wave_gemm<1, 0>(sX, EMB, sWt2, 64, bw.mlp1_b, 16, sM, MLPH, nullptr, 0);
    __syncthreads();
    load_wt(bw.mlp2_w, 256, 64, sWt2);
    __syncthreads();
    wave_gemm<0, 1>(sM, MLPH, sWt2, 256, bw.mlp2_b, 4, nullptr, 0, sH, EMB);
    __syncthreads();
  }

  // ---------------- decoders: sigmoid(h W1+b) -> sigmoid(. W2+b) ---------
  for (int t = tid; t < LPAD; t += 256) {
#pragma unroll
    for (int e = 0; e < EMB; ++e)
      sX[t * EMB + e] = (t < LSEQ) ? (h16)sH[t * EMB + e] : (h16)0.f;
  }
  __syncthreads();
  load_wt(P.dec1_w, 64, 64, sScr);
  __syncthreads();
  h16* sD = sBig;                           // [304][64] dec1 output
  wave_gemm<2, 0>(sX, EMB, sScr, 64, P.dec1_b, 4, sD, EMB, nullptr, 0);
  __syncthreads();
  load_wt(P.dec2_w, 64, 64, sScr);
  __syncthreads();
  wave_gemm<2, 0>(sD, EMB, sScr, 64, P.dec2_b, 4, sX, EMB, nullptr, 0);
  __syncthreads();

  // ---------------- 2x conv3x3(SAME)+sigmoid on 8x8 tiles, then paste ----
  float w1[9], w2[9];
#pragma unroll
  for (int i = 0; i < 9; ++i) { w1[i] = P.conv1_w[i]; w2[i] = P.conv2_w[i]; }
  const float b1 = P.conv1_b[0], b2 = P.conv2_b[0];

  for (int t = tid; t < LSEQ; t += 256) {
    float a[64], c1[64];
#pragma unroll
    for (int i = 0; i < 64; ++i) a[i] = (float)sX[t * EMB + i];
#pragma unroll
    for (int i = 0; i < 8; ++i)
#pragma unroll
      for (int j = 0; j < 8; ++j) {
        float acc = b1;
#pragma unroll
        for (int di = -1; di <= 1; ++di)
#pragma unroll
          for (int dj = -1; dj <= 1; ++dj) {
            int ii = i + di, jj = j + dj;
            if (ii >= 0 && ii < 8 && jj >= 0 && jj < 8)
              acc += a[ii * 8 + jj] * w1[(di + 1) * 3 + (dj + 1)];
          }
        c1[i * 8 + j] = sigmoid_f(acc);
      }
    const int r = t / 15, cc = t - r * 15;
    float* ob = out + (long)b * 19200 + (r * 8) * 120 + cc * 8;
#pragma unroll
    for (int i = 0; i < 8; ++i)
#pragma unroll
      for (int j = 0; j < 8; ++j) {
        float acc = b2;
#pragma unroll
        for (int di = -1; di <= 1; ++di)
#pragma unroll
          for (int dj = -1; dj <= 1; ++dj) {
            int ii = i + di, jj = j + dj;
            if (ii >= 0 && ii < 8 && jj >= 0 && jj < 8)
              acc += c1[ii * 8 + jj] * w2[(di + 1) * 3 + (dj + 1)];
          }
        ob[i * 120 + j] = sigmoid_f(acc);
      }
  }
}

// ---------------------------------------------------------------- launch
extern "C" void kernel_launch(void* const* d_in, const int* in_sizes, int n_in,
                              void* d_out, int out_size, void* d_ws, size_t ws_size,
                              hipStream_t stream) {
  (void)in_sizes; (void)n_in; (void)out_size; (void)d_ws; (void)ws_size;
  NetW P;
  P.x       = (const float*)d_in[0];
  P.fc_w    = (const float*)d_in[1];
  P.fc_b    = (const float*)d_in[2];
  P.fcln_g  = (const float*)d_in[3];
  P.fcln_b  = (const float*)d_in[4];
  P.pos     = (const float*)d_in[5];
  P.dec1_w  = (const float*)d_in[6];
  P.dec1_b  = (const float*)d_in[7];
  P.dec2_w  = (const float*)d_in[8];
  P.dec2_b  = (const float*)d_in[9];
  P.conv1_w = (const float*)d_in[10];
  P.conv1_b = (const float*)d_in[11];
  P.conv2_w = (const float*)d_in[12];
  P.conv2_b = (const float*)d_in[13];
  for (int i = 0; i < 3; ++i) {
    const int s = 14 + 12 * i;
    P.blk[i].ln1_g  = (const float*)d_in[s + 0];
    P.blk[i].ln1_b  = (const float*)d_in[s + 1];
    P.blk[i].qkv_w  = (const float*)d_in[s + 2];
    P.blk[i].qkv_b  = (const float*)d_in[s + 3];
    P.blk[i].proj_w = (const float*)d_in[s + 4];
    P.blk[i].proj_b = (const float*)d_in[s + 5];
    P.blk[i].ln2_g  = (const float*)d_in[s + 6];
    P.blk[i].ln2_b  = (const float*)d_in[s + 7];
    P.blk[i].mlp1_w = (const float*)d_in[s + 8];
    P.blk[i].mlp1_b = (const float*)d_in[s + 9];
    P.blk[i].mlp2_w = (const float*)d_in[s + 10];
    P.blk[i].mlp2_b = (const float*)d_in[s + 11];
  }
  decoder_fused<<<256, 256, 0, stream>>>(P, (float*)d_out);
}
